// BertCls_64372969832471
// MI455X (gfx1250) — matrix operand order repaired and checked by reading, not compiled, not run on hardware
//
#include <hip/hip_runtime.h>
#include <math.h>
#include <stdint.h>

// ---------------------------------------------------------------------------
// MI455X (gfx1250) Longformer-style BERT classifier forward.
// All GEMM-shaped math -> v_wmma_f32_16x16x32_f16 (wave32 WMMA).
// Tile staging uses CDNA5 GLOBAL_LOAD_ASYNC_TO_LDS_B128 (ASYNCcnt) so data
// never round-trips through VGPRs. Weights pre-converted once per launch to
// f16 [N,K]. Banded attention = 2 WMMA GEMMs + fused masked softmax in
// ~205KB of the 320KB WGP LDS.
// ---------------------------------------------------------------------------

typedef _Float16 h16;
typedef __attribute__((ext_vector_type(8)))  _Float16 hv8;
typedef __attribute__((ext_vector_type(16))) _Float16 v16h;
typedef __attribute__((ext_vector_type(8)))  float    v8f;

#define SEQ   2048
#define BB    2
#define DIM   256
#define NH    8
#define DH    32
#define NL    12
#define FF    1024
#define WIN   128
#define NCHK  (SEQ / WIN)     // 16
#define ROWS  (BB * SEQ)      // 4096
#define NEGV  (-1.0e9f)
#define INV_SQRT_DH 0.17677669529663687f

// CDNA5 async memory->LDS copy (16B per lane), tracked by ASYNCcnt.
// VDST operand carries the LDS byte address (low 32 bits of generic pointer).
__device__ __forceinline__ void async_ld_b128(void* lds, const void* gaddr)
{
    unsigned l = (unsigned)(uintptr_t)lds;
    unsigned long long g = (unsigned long long)(uintptr_t)gaddr;
    asm volatile("global_load_async_to_lds_b128 %0, %1, off"
                 :: "v"(l), "v"(g) : "memory");
}
__device__ __forceinline__ void wait_async()
{
    asm volatile("s_wait_asynccnt 0" ::: "memory");
}

// Fragment maps for 16x16x32 f16 WMMA (wave32), CDNA5 ISA 7.12.2:
//  A (16x32): lane L: m=L%16, half=L/16; elem j -> k=16*half+j (this file's order)
//  B (32x16): lane L: n=L%16, half=L/16; elem j -> k=16*half+j (this file's order, as A; not the ISA table)
//  C/D f32  : lane L: n=L%16; elem r -> m=r+8*(L/16)

// ================= weight convert+transpose: f32[K,N] -> f16[N,K] ==========
__global__ void wt_convert_kernel(const float* __restrict__ W, h16* __restrict__ Wt,
                                  int K, int N)
{
    __shared__ float tile[32][33];
    int l = blockIdx.z;
    const float* Wl = W + (size_t)l * K * N;
    h16* Wtl = Wt + (size_t)l * K * N;
    int n0 = blockIdx.x * 32, k0 = blockIdx.y * 32;
    int t = threadIdx.x;
    int tn = t & 31, tk = t >> 5;             // 32 x 8
    #pragma unroll
    for (int i = 0; i < 4; ++i)               // coalesced along n
        tile[tk + i * 8][tn] = Wl[(size_t)(k0 + tk + i * 8) * N + n0 + tn];
    __syncthreads();
    #pragma unroll
    for (int i = 0; i < 4; ++i)               // coalesced along k
        Wtl[(size_t)(n0 + tk + i * 8) * K + k0 + tn] = (h16)tile[tn][tk + i * 8];
}

// ============================ embedding + LN ===============================
__global__ void embed_ln_kernel(const int* __restrict__ ids,
                                const float* __restrict__ wemb,
                                const float* __restrict__ pemb,
                                const float* __restrict__ gs,
                                const float* __restrict__ gb,
                                float* __restrict__ xo, h16* __restrict__ xh)
{
    __shared__ float red[256];
    __shared__ float stat[2];
    int row = blockIdx.x, t = threadIdx.x;
    int s = row % SEQ;
    int tok = ids[row];
    float v = wemb[(size_t)tok * DIM + t] + pemb[(size_t)(s + 2) * DIM + t];
    red[t] = v; __syncthreads();
    for (int off = 128; off > 0; off >>= 1) { if (t < off) red[t] += red[t + off]; __syncthreads(); }
    if (t == 0) stat[0] = red[0] * (1.0f / 256.0f);
    __syncthreads();
    float mu = stat[0];
    float d = v - mu;
    red[t] = d * d; __syncthreads();
    for (int off = 128; off > 0; off >>= 1) { if (t < off) red[t] += red[t + off]; __syncthreads(); }
    if (t == 0) stat[1] = red[0] * (1.0f / 256.0f);
    __syncthreads();
    float y = d * rsqrtf(stat[1] + 1e-5f) * gs[t] + gb[t];
    xo[(size_t)row * DIM + t] = y;
    xh[(size_t)row * DIM + t] = (h16)y;
}

// ======================= residual add + LayerNorm ==========================
__global__ void add_ln_kernel(float* __restrict__ x, const float* __restrict__ addv,
                              const float* __restrict__ gs, const float* __restrict__ gb,
                              h16* __restrict__ xh)
{
    __shared__ float red[256];
    __shared__ float stat[2];
    int row = blockIdx.x, t = threadIdx.x;
    float v = x[(size_t)row * DIM + t] + addv[(size_t)row * DIM + t];
    red[t] = v; __syncthreads();
    for (int off = 128; off > 0; off >>= 1) { if (t < off) red[t] += red[t + off]; __syncthreads(); }
    if (t == 0) stat[0] = red[0] * (1.0f / 256.0f);
    __syncthreads();
    float mu = stat[0];
    float d = v - mu;
    red[t] = d * d; __syncthreads();
    for (int off = 128; off > 0; off >>= 1) { if (t < off) red[t] += red[t + off]; __syncthreads(); }
    if (t == 0) stat[1] = red[0] * (1.0f / 256.0f);
    __syncthreads();
    float y = d * rsqrtf(stat[1] + 1e-5f) * gs[t] + gb[t];
    x[(size_t)row * DIM + t]  = y;
    xh[(size_t)row * DIM + t] = (h16)y;
}

// ========================= WMMA GEMM  C = A*Bt^T + bias ====================
// A:  f16 [M,K] row-major.   Bt: f16 [N,K] row-major (pre-transposed weight).
// Block = 256 threads (8 waves), tile 128(M) x 128(N), K-step 32.
// Waves 4(M) x 2(N); each wave 32x64 => 8 WMMAs per K-step.
// Staging via GLOBAL_LOAD_ASYNC_TO_LDS_B128 (no VGPR round trip).
// Requires M%128==0, N%128==0, K%32==0 (true for all call sites).
__global__ void gemm_h16_kernel(const h16* __restrict__ A, const h16* __restrict__ Bt,
                                const float* __restrict__ bias, float scale, int gelu,
                                int M, int N, int K,
                                float* __restrict__ outF, h16* __restrict__ outH)
{
    __shared__ h16 As[128][40];   // [m][k] tile
    __shared__ h16 Bs[128][40];   // [n][k] tile
    int bm = blockIdx.y * 128;
    int bn = blockIdx.x * 128;
    int tid = threadIdx.x;
    int lane = tid & 31, wave = tid >> 5;
    int hl = lane >> 4, lm = lane & 15;
    int wm = (wave >> 1) * 32;          // 0,32,64,96
    int wn = (wave & 1) * 64;           // 0,64
    v8f acc[2][4] = {};

    for (int k0 = 0; k0 < K; k0 += 32) {
        // async-stage A & B tiles: 128x32 f16 each, 16B per copy
        #pragma unroll
        for (int it = 0; it < 2; ++it) {
            int slot = tid + it * 256;          // 512 slots of 8 f16
            int r = slot >> 2, cv = (slot & 3) * 8;
            async_ld_b128(&As[r][cv], A  + (size_t)(bm + r) * K + k0 + cv);
            async_ld_b128(&Bs[r][cv], Bt + (size_t)(bn + r) * K + k0 + cv);
        }
        if (k0 + 32 < K)   // near-temporal prefetch of next weight tile
            __builtin_prefetch(Bt + (size_t)bn * K + k0 + 32, 0, 3);
        wait_async();
        __syncthreads();

        #pragma unroll
        for (int mi = 0; mi < 2; ++mi) {
            v16h a;
            #pragma unroll
            for (int j = 0; j < 16; ++j) {
                int kkA = 16 * hl + j;
                a[j] = As[wm + mi * 16 + lm][kkA];
            }
            #pragma unroll
            for (int ni = 0; ni < 4; ++ni) {
                v16h b;
                #pragma unroll
                for (int j = 0; j < 16; ++j)
                    b[j] = Bs[wn + ni * 16 + lm][16 * hl + j];
                acc[mi][ni] = __builtin_amdgcn_wmma_f32_16x16x32_f16(
                    false, a, false, b, (short)0, acc[mi][ni], false, false);
            }
        }
        __syncthreads();
    }

    #pragma unroll
    for (int mi = 0; mi < 2; ++mi)
        #pragma unroll
        for (int ni = 0; ni < 4; ++ni)
            #pragma unroll
            for (int r = 0; r < 8; ++r) {
                int m = bm + wm + mi * 16 + r + 8 * hl;
                int n = bn + wn + ni * 16 + lm;
                float v = acc[mi][ni][r];
                if (bias) v += bias[n];
                v *= scale;
                if (gelu) v = 0.5f * v * (1.0f + erff(v * 0.70710678118f));
                if (outF) outF[(size_t)m * N + n] = v;
                if (outH) outH[(size_t)m * N + n] = (h16)v;
            }
}

// ===================== banded local attention (WMMA) =======================
// One block = 64 queries of one (b, head, chunk): scores QK^T (K=32, 1 WMMA
// per tile) -> masked softmax in LDS -> probs@V (K=384, 12 WMMAs per tile).
__global__ void attn_local_kernel(const h16* __restrict__ qh, const h16* __restrict__ kh,
                                  const h16* __restrict__ vh, const int* __restrict__ amask,
                                  h16* __restrict__ attn)
{
    extern __shared__ char smem[];
    h16 (*qs)[32] = (h16(*)[32])(smem);                       //  4096 B
    h16 (*ks)[32] = (h16(*)[32])(smem + 4096);                // 24576 B
    h16 (*vs)[32] = (h16(*)[32])(smem + 28672);               // 24576 B
    float* scores = (float*)(smem + 53248);                   // 64*388*4 = 99328 B
    h16*   probs  = (h16*)  (smem + 152576);                  // 64*392*2 = 50176 B
    float* gsc    = (float*)(smem + 202752);                  // 256 B
    float* gpr    = (float*)(smem + 203008);                  // 256 B
    float* k0s    = (float*)(smem + 203264);                  // 128 B
    float* v0s    = (float*)(smem + 203392);                  // 128 B
    int*   kval   = (int*)  (smem + 203520);                  // 1536 B (tot 205056)
    __shared__ float redm[64][4];
    __shared__ float reds[64][4];

    int bid = blockIdx.x;
    int qhalf = bid & 1;
    int rest = bid >> 1;
    int c = rest % NCHK; rest /= NCHK;
    int hd = rest & 7;
    int b  = rest >> 3;

    int tid = threadIdx.x;
    int q0 = c * WIN + qhalf * 64;
    int kbase = c * WIN - WIN;

    // ---- async-stage q / k / v (16B per lane) ----
    {
        int r = tid >> 2, sv = (tid & 3) * 8;   // 256 tasks: 64 rows x 4 vecs
        async_ld_b128(&qs[r][sv],
                      qh + ((size_t)(b * SEQ + q0 + r)) * DIM + hd * DH + sv);
    }
    #pragma unroll
    for (int it = 0; it < 6; ++it) {            // 1536 tasks: 384 rows x 4 vecs
        int task = tid + it * 256;
        int r = task >> 2, sv = (task & 3) * 8;
        int kp = kbase + r;
        if (kp >= 0 && kp < SEQ) {
            size_t off = ((size_t)(b * SEQ + kp)) * DIM + hd * DH + sv;
            async_ld_b128(&ks[r][sv], kh + off);
            async_ld_b128(&vs[r][sv], vh + off);
        } else {
            hv8 z = {};
            *(hv8*)&ks[r][sv] = z;
            *(hv8*)&vs[r][sv] = z;
        }
    }
    for (int r = tid; r < 384; r += 256) {
        int kp = kbase + r;
        kval[r] = (kp >= 0 && kp < SEQ && kp != 0 && amask[b * SEQ + kp] > 0) ? 1 : 0;
    }
    if (tid < 32) {
        size_t off0 = ((size_t)(b * SEQ)) * DIM + hd * DH + tid;
        k0s[tid] = (float)kh[off0];
        v0s[tid] = (float)vh[off0];
    }
    wait_async();
    __syncthreads();

    int lane = tid & 31, wave = tid >> 5;
    int hl = lane >> 4, lm = lane & 15;

    // ---- scores = q @ k^T : 4x24 tiles, K=DH=32 -> 1 WMMA each ----
    for (int i = 0; i < 12; ++i) {
        int t  = wave * 12 + i;
        int mt = t / 24, nt = t % 24;
        v16h a, bfr;
        #pragma unroll
        for (int j = 0; j < 16; ++j) {
            int kkA = 16 * hl + j;
            a[j]   = qs[mt * 16 + lm][kkA];
            bfr[j] = ks[nt * 16 + lm][16 * hl + j];
        }
        v8f accv = {};
        accv = __builtin_amdgcn_wmma_f32_16x16x32_f16(false, a, false, bfr,
                                                      (short)0, accv, false, false);
        #pragma unroll
        for (int r = 0; r < 8; ++r) {
            int m = mt * 16 + r + 8 * hl;
            int n = nt * 16 + lm;
            int kp = kbase + n;
            int qp = q0 + m;
            bool ok = kval[n] && (kp - qp <= WIN) && (qp - kp <= WIN);
            scores[m * 388 + n] = ok ? accv[r] : NEGV;
        }
    }
    if (tid < 64) {   // global-key score (softmax column 0)
        float acc = 0.0f;
        for (int d = 0; d < 32; ++d) acc += (float)qs[tid][d] * k0s[d];
        gsc[tid] = acc;
    }
    __syncthreads();

    // ---- fused masked softmax: 4 threads per query row ----
    {
        int row = tid >> 2, part = tid & 3;
        float mx = -3.4e38f;
        for (int n = part; n < 384; n += 4) mx = fmaxf(mx, scores[row * 388 + n]);
        if (part == 0) mx = fmaxf(mx, gsc[row]);
        redm[row][part] = mx;
        __syncthreads();
        mx = fmaxf(fmaxf(redm[row][0], redm[row][1]),
                   fmaxf(redm[row][2], redm[row][3]));
        float sum = 0.0f;
        for (int n = part; n < 384; n += 4) {
            float e = expf(scores[row * 388 + n] - mx);
            probs[row * 392 + n] = (h16)e;
            sum += e;
        }
        if (part == 0) sum += expf(gsc[row] - mx);
        reds[row][part] = sum;
        __syncthreads();
        float inv = 1.0f / (reds[row][0] + reds[row][1] + reds[row][2] + reds[row][3]);
        if (part == 0) gpr[row] = expf(gsc[row] - mx) * inv;
        for (int n = part; n < 384; n += 4)
            probs[row * 392 + n] = (h16)((float)probs[row * 392 + n] * inv);
    }
    __syncthreads();

    // ---- out = probs @ v : 4x2 tiles, K=384 -> 12 WMMAs each ----
    {
        int mt = wave >> 1, nt = wave & 1;
        v8f accv = {};
        for (int ks0 = 0; ks0 < 384; ks0 += 32) {
            v16h a, bfr;
            #pragma unroll
            for (int j = 0; j < 16; ++j) {
                int kkA = 16 * hl + j;
                a[j]   = probs[(mt * 16 + lm) * 392 + ks0 + kkA];
                bfr[j] = vs[ks0 + 16 * hl + j][nt * 16 + lm];
            }
            accv = __builtin_amdgcn_wmma_f32_16x16x32_f16(false, a, false, bfr,
                                                          (short)0, accv, false, false);
        }
        #pragma unroll
        for (int r = 0; r < 8; ++r) {
            int m = mt * 16 + r + 8 * hl;
            int n = nt * 16 + lm;
            float val = accv[r] + gpr[m] * v0s[n];
            attn[((size_t)(b * SEQ + q0 + m)) * DIM + hd * DH + n] = (h16)val;
        }
    }
}

// ====================== global-token attention (s==0) ======================
__global__ void attn_global_kernel(const float* __restrict__ x, const float* __restrict__ Wqg,
                                   const float* __restrict__ bqg, const h16* __restrict__ kg,
                                   const h16* __restrict__ vg, const int* __restrict__ amask,
                                   h16* __restrict__ attn)
{
    __shared__ float qgs[32];
    __shared__ float gsb[SEQ];
    __shared__ float red[256];
    __shared__ float part[8][32];
    int b = blockIdx.x >> 3, hd = blockIdx.x & 7;
    int t = threadIdx.x;

    if (t < 32) {
        float acc = bqg[hd * DH + t];
        const float* xr = x + (size_t)(b * SEQ) * DIM;
        for (int j = 0; j < DIM; ++j) acc += xr[j] * Wqg[j * DIM + hd * DH + t];
        qgs[t] = acc * INV_SQRT_DH;
    }
    __syncthreads();
    for (int s = t; s < SEQ; s += 256) {
        float acc = 0.0f;
        const h16* kr = kg + ((size_t)(b * SEQ + s)) * DIM + hd * DH;
        for (int d = 0; d < 32; ++d) acc += qgs[d] * (float)kr[d];
        gsb[s] = (amask[b * SEQ + s] > 0) ? acc : NEGV;
    }
    __syncthreads();
    float mx = -3.4e38f;
    for (int s = t; s < SEQ; s += 256) mx = fmaxf(mx, gsb[s]);
    red[t] = mx; __syncthreads();
    for (int off = 128; off > 0; off >>= 1) { if (t < off) red[t] = fmaxf(red[t], red[t + off]); __syncthreads(); }
    mx = red[0]; __syncthreads();
    float sm = 0.0f;
    for (int s = t; s < SEQ; s += 256) { float e = expf(gsb[s] - mx); gsb[s] = e; sm += e; }
    red[t] = sm; __syncthreads();
    for (int off = 128; off > 0; off >>= 1) { if (t < off) red[t] += red[t + off]; __syncthreads(); }
    float inv = 1.0f / red[0];
    __syncthreads();
    {
        int n = t & 31, ch = t >> 5;
        float acc = 0.0f;
        for (int s = ch; s < SEQ; s += 8)
            acc += gsb[s] * inv * (float)vg[((size_t)(b * SEQ + s)) * DIM + hd * DH + n];
        part[ch][n] = acc;
    }
    __syncthreads();
    if (t < 32) {
        float o = 0.0f;
        for (int ch = 0; ch < 8; ++ch) o += part[ch][t];
        attn[((size_t)(b * SEQ)) * DIM + hd * DH + t] = (h16)o;
    }
}

// ============================== pooler + head ==============================
__global__ void pool_kernel(const float* __restrict__ x, const float* __restrict__ Wp,
                            const float* __restrict__ bp, float* __restrict__ pooled)
{
    int b = blockIdx.x, n = threadIdx.x;
    float acc = bp[n];
    const float* xr = x + (size_t)(b * SEQ) * DIM;
    for (int j = 0; j < DIM; ++j) acc += xr[j] * Wp[j * DIM + n];
    pooled[b * DIM + n] = tanhf(acc);
}

__global__ void head_kernel(const float* __restrict__ pooled, const float* __restrict__ numeric,
                            const float* __restrict__ Wfc, const float* __restrict__ bfc,
                            float* __restrict__ out)
{
    int t = threadIdx.x;
    if (t < BB) {
        float acc = bfc[0];
        for (int j = 0; j < DIM; ++j) acc += pooled[t * DIM + j] * Wfc[j];
        for (int i = 0; i < 16; ++i)  acc += numeric[t * 16 + i] * Wfc[DIM + i];
        out[t] = 1.0f / (1.0f + expf(-acc));
    }
}

// ============================== host driver ================================
extern "C" void kernel_launch(void* const* d_in, const int* in_sizes, int n_in,
                              void* d_out, int out_size, void* d_ws, size_t ws_size,
                              hipStream_t stream)
{
    (void)in_sizes; (void)n_in; (void)out_size; (void)ws_size;
    const int*   input_ids = (const int*)  d_in[0];
    const int*   amask     = (const int*)  d_in[1];
    const float* numeric   = (const float*)d_in[2];
    const float* word_emb  = (const float*)d_in[3];
    const float* pos_emb   = (const float*)d_in[4];
    const float* ln_e_s    = (const float*)d_in[5];
    const float* ln_e_b    = (const float*)d_in[6];
    const float* Wp        = (const float*)d_in[7];
    const float* bp        = (const float*)d_in[8];
    const float* Wfc       = (const float*)d_in[9];
    const float* bfc       = (const float*)d_in[10];
    const float* Wi        = (const float*)d_in[11];
    const float* bi        = (const float*)d_in[12];
    const float* Wf        = (const float*)d_in[13];
    const float* bf        = (const float*)d_in[14];
    const float* Wq        = (const float*)d_in[15];
    const float* Wk        = (const float*)d_in[16];
    const float* Wv        = (const float*)d_in[17];
    const float* Wo        = (const float*)d_in[18];
    const float* Wqg       = (const float*)d_in[19];
    const float* Wkg       = (const float*)d_in[20];
    const float* Wvg       = (const float*)d_in[21];
    const float* bq        = (const float*)d_in[22];
    const float* bk        = (const float*)d_in[23];
    const float* bv        = (const float*)d_in[24];
    const float* bo        = (const float*)d_in[25];
    const float* bqg       = (const float*)d_in[26];
    const float* bkg       = (const float*)d_in[27];
    const float* bvg       = (const float*)d_in[28];
    const float* ln1_s     = (const float*)d_in[29];
    const float* ln2_s     = (const float*)d_in[30];
    const float* ln1_b     = (const float*)d_in[31];
    const float* ln2_b     = (const float*)d_in[32];

    // -------- workspace carve (~55 MB) --------
    char* wsp = (char*)d_ws;
    auto carve = [&](size_t bytes) { char* r = wsp; wsp += (bytes + 255) & ~(size_t)255; return r; };
    float* x      = (float*)carve((size_t)ROWS * DIM * 4);
    h16*   xh     = (h16*)  carve((size_t)ROWS * DIM * 2);
    h16*   qhb    = (h16*)  carve((size_t)ROWS * DIM * 2);
    h16*   khb    = (h16*)  carve((size_t)ROWS * DIM * 2);
    h16*   vhb    = (h16*)  carve((size_t)ROWS * DIM * 2);
    h16*   kghb   = (h16*)  carve((size_t)ROWS * DIM * 2);
    h16*   vghb   = (h16*)  carve((size_t)ROWS * DIM * 2);
    h16*   attnh  = (h16*)  carve((size_t)ROWS * DIM * 2);
    h16*   ffnh   = (h16*)  carve((size_t)ROWS * FF * 2);
    float* tmp    = (float*)carve((size_t)ROWS * DIM * 4);
    float* pooled = (float*)carve((size_t)BB * DIM * 4);
    // transposed f16 weights [N,K] per layer
    h16* wtq  = (h16*)carve((size_t)NL * DIM * DIM * 2);
    h16* wtk  = (h16*)carve((size_t)NL * DIM * DIM * 2);
    h16* wtv  = (h16*)carve((size_t)NL * DIM * DIM * 2);
    h16* wto  = (h16*)carve((size_t)NL * DIM * DIM * 2);
    h16* wtkg = (h16*)carve((size_t)NL * DIM * DIM * 2);
    h16* wtvg = (h16*)carve((size_t)NL * DIM * DIM * 2);
    h16* wti  = (h16*)carve((size_t)NL * DIM * FF * 2);
    h16* wtf  = (h16*)carve((size_t)NL * FF * DIM * 2);

    const dim3 blk(256);
    const dim3 gDD(DIM / 32, DIM / 32, NL);
    wt_convert_kernel<<<gDD, blk, 0, stream>>>(Wq,  wtq,  DIM, DIM);
    wt_convert_kernel<<<gDD, blk, 0, stream>>>(Wk,  wtk,  DIM, DIM);
    wt_convert_kernel<<<gDD, blk, 0, stream>>>(Wv,  wtv,  DIM, DIM);
    wt_convert_kernel<<<gDD, blk, 0, stream>>>(Wo,  wto,  DIM, DIM);
    wt_convert_kernel<<<gDD, blk, 0, stream>>>(Wkg, wtkg, DIM, DIM);
    wt_convert_kernel<<<gDD, blk, 0, stream>>>(Wvg, wtvg, DIM, DIM);
    wt_convert_kernel<<<dim3(FF / 32, DIM / 32, NL), blk, 0, stream>>>(Wi, wti, DIM, FF);
    wt_convert_kernel<<<dim3(DIM / 32, FF / 32, NL), blk, 0, stream>>>(Wf, wtf, FF, DIM);

    const dim3 gP(DIM / 128, ROWS / 128);   // N=256 GEMMs
    const dim3 gI(FF / 128,  ROWS / 128);   // N=1024 GEMM
    const size_t ATTN_SMEM = 205056;

    embed_ln_kernel<<<ROWS, blk, 0, stream>>>(input_ids, word_emb, pos_emb,
                                              ln_e_s, ln_e_b, x, xh);

    for (int l = 0; l < NL; ++l) {
        const size_t oDD = (size_t)l * DIM * DIM;
        const size_t oDF = (size_t)l * DIM * FF;
        const float* bq_l  = bq  + l * DIM;  const float* bk_l  = bk  + l * DIM;
        const float* bv_l  = bv  + l * DIM;  const float* bo_l  = bo  + l * DIM;
        const float* bqg_l = bqg + l * DIM;  const float* bkg_l = bkg + l * DIM;
        const float* bvg_l = bvg + l * DIM;
        const float* bi_l  = bi  + l * FF;   const float* bf_l  = bf  + l * DIM;

        gemm_h16_kernel<<<gP, blk, 0, stream>>>(xh, wtq + oDD, bq_l, INV_SQRT_DH, 0,
                                                ROWS, DIM, DIM, nullptr, qhb);
        gemm_h16_kernel<<<gP, blk, 0, stream>>>(xh, wtk + oDD, bk_l, 1.0f, 0,
                                                ROWS, DIM, DIM, nullptr, khb);
        gemm_h16_kernel<<<gP, blk, 0, stream>>>(xh, wtv + oDD, bv_l, 1.0f, 0,
                                                ROWS, DIM, DIM, nullptr, vhb);
        gemm_h16_kernel<<<gP, blk, 0, stream>>>(xh, wtkg + oDD, bkg_l, 1.0f, 0,
                                                ROWS, DIM, DIM, nullptr, kghb);
        gemm_h16_kernel<<<gP, blk, 0, stream>>>(xh, wtvg + oDD, bvg_l, 1.0f, 0,
                                                ROWS, DIM, DIM, nullptr, vghb);

        attn_local_kernel<<<BB * NH * NCHK * 2, blk, ATTN_SMEM, stream>>>(
            qhb, khb, vhb, amask, attnh);
        attn_global_kernel<<<BB * NH, blk, 0, stream>>>(
            x, Wqg + oDD, bqg_l, kghb, vghb, amask, attnh);

        gemm_h16_kernel<<<gP, blk, 0, stream>>>(attnh, wto + oDD, bo_l, 1.0f, 0,
                                                ROWS, DIM, DIM, tmp, nullptr);
        add_ln_kernel<<<ROWS, blk, 0, stream>>>(x, tmp, ln1_s + l * DIM,
                                                ln1_b + l * DIM, xh);

        gemm_h16_kernel<<<gI, blk, 0, stream>>>(xh, wti + oDF, bi_l, 1.0f, 1,
                                                ROWS, FF, DIM, nullptr, ffnh);
        gemm_h16_kernel<<<gP, blk, 0, stream>>>(ffnh, wtf + oDF, bf_l, 1.0f, 0,
                                                ROWS, DIM, FF, tmp, nullptr);
        add_ln_kernel<<<ROWS, blk, 0, stream>>>(x, tmp, ln2_s + l * DIM,
                                                ln2_b + l * DIM, xh);
    }

    pool_kernel<<<BB, blk, 0, stream>>>(x, Wp, bp, pooled);
    head_kernel<<<1, 64, 0, stream>>>(pooled, numeric, Wfc, bfc, (float*)d_out);
}
